// EffortMoeDetector_23356032155699
// MI455X (gfx1250) — matrix operand order repaired and checked by reading, not compiled, not run on hardware
//
#include <hip/hip_runtime.h>
#include <hip/hip_bf16.h>

// Problem sizes (fixed by the reference)
#define NTOK  4096
#define DIN   1024
#define DOUT  1024
#define NE    8
#define RK    16
#define HGATE 256

typedef __bf16 bf16_t;
typedef __attribute__((ext_vector_type(16))) __bf16        v16bf;
typedef __attribute__((ext_vector_type(8)))  float         v8f;
typedef __attribute__((ext_vector_type(4)))  unsigned int  v4u;
typedef __attribute__((ext_vector_type(4)))  float         v4f;

// One WMMA operand fragment: 16 bf16 per lane = 32 bytes = two 16B chunks.
union Frag {
    v16bf v;
    v4u   u[2];
};

// Fragment set for a 32x64 wave tile k-step: 2 A tiles, 4 B tiles.
struct Frags {
    Frag a[2];
    Frag b[4];
};

__device__ __forceinline__ v8f wmma_bf16(const Frag& a, const Frag& b, v8f c) {
    // D = A(16x32) * B(32x16) + C, fp32 accumulate
    return __builtin_amdgcn_wmma_f32_16x16x32_bf16(
        /*neg_a=*/false, a.v, /*neg_b=*/false, b.v,
        /*c_mod=*/(short)0, c, /*reuse_a=*/false, /*reuse_b=*/false);
}

// Per-lane fragment layouts per CDNA5 ISA 7.12.2:
//   A 16x32 bf16: row = lane&15; lane<16 holds K {0..7, 16..23},
//                 lane>=16 holds K {8..15, 24..31}  -> chunks at (h*8, h*8+16)
//   B 32x16 bf16: col = lane&15; same K split as A
//                 (lane<16: {0..7, 16..23}, lane>=16: {8..15, 24..31})
__device__ __forceinline__ void load_frags(Frags& f,
                                           const bf16_t* __restrict__ ap,
                                           const bf16_t* __restrict__ bp,
                                           size_t row16) {
    f.a[0].u[0] = *(const v4u*)(ap);
    f.a[0].u[1] = *(const v4u*)(ap + 16);
    f.a[1].u[0] = *(const v4u*)(ap + row16);
    f.a[1].u[1] = *(const v4u*)(ap + row16 + 16);
    #pragma unroll
    for (int j = 0; j < 4; ++j) {
        f.b[j].u[0] = *(const v4u*)(bp + (size_t)j * row16);
        f.b[j].u[1] = *(const v4u*)(bp + (size_t)j * row16 + 16);
    }
}

__device__ __forceinline__ void mma8(v8f c[2][4], const Frags& f) {
    #pragma unroll
    for (int i = 0; i < 2; ++i)
        #pragma unroll
        for (int j = 0; j < 4; ++j)
            c[i][j] = wmma_bf16(f.a[i], f.b[j], c[i][j]);
}

// 32x64 wave tile (2x4 WMMA 16x16 tiles), two-deep software pipeline over K
// in steps of 64. A: row-major M x K. B: row-major N x K (so the math-B column
// is contiguous along K).
__device__ __forceinline__ void mma_wave_tile(const bf16_t* __restrict__ A,
                                              const bf16_t* __restrict__ B,
                                              int K, int m0, int n0, int lane,
                                              v8f c[2][4]) {
    const int r = lane & 15;
    const int h = lane >> 4;
    const bf16_t* ap = A + (size_t)(m0 + r) * K + (h << 3);
    const bf16_t* bp = B + (size_t)(n0 + r) * K + (h << 3);
    const size_t row16 = (size_t)16 * K;

    Frags f0, f1;
    load_frags(f0, ap, bp, row16);
    for (int k0 = 0; k0 < K; k0 += 64) {
        load_frags(f1, ap + 32, bp + 32, row16);      // prefetch next half-step
        __builtin_prefetch(ap + 64, 0, 3);            // near-scope prefetch
        __builtin_prefetch(bp + 64, 0, 3);
        mma8(c, f0);                                  // overlaps f1 loads
        if (k0 + 64 < K)
            load_frags(f0, ap + 64, bp + 64, row16);  // prefetch next step
        mma8(c, f1);                                  // overlaps f0 loads
        ap += 64;
        bp += 64;
    }
}

// ---------------------------------------------------------------------------
// Kernel 1: fp32 -> bf16 conversions (x, weight_main) and transposed Wg1.
// ---------------------------------------------------------------------------
__global__ void convert_kernel(const float* __restrict__ x,
                               const float* __restrict__ wmain,
                               const float* __restrict__ wg1,
                               bf16_t* __restrict__ xb,
                               bf16_t* __restrict__ wmb,
                               bf16_t* __restrict__ wg1t) {
    const int idx = blockIdx.x * blockDim.x + threadIdx.x;
    const int stride = gridDim.x * blockDim.x;
    for (int i = idx; i < NTOK * DIN; i += stride) xb[i] = (bf16_t)x[i];
    for (int i = idx; i < DOUT * DIN; i += stride) wmb[i] = (bf16_t)wmain[i];
    for (int i = idx; i < DIN * HGATE; i += stride) {
        const int rr = i / HGATE;          // row in (IN, HG)
        const int cc = i - rr * HGATE;     // col
        wg1t[(size_t)cc * DIN + rr] = (bf16_t)wg1[i];   // (HG, IN)
    }
}

// ---------------------------------------------------------------------------
// Kernel 2: h = relu(x @ Wg1 + bg1), bf16 WMMA GEMM 4096x256x1024.
// Block tile 128x128, 8 waves (4x2), wave tile 32x64.
// ---------------------------------------------------------------------------
__global__ __launch_bounds__(256) void gate_hidden_gemm(
        const bf16_t* __restrict__ xb, const bf16_t* __restrict__ wg1t,
        const float* __restrict__ bg1, bf16_t* __restrict__ h) {
    const int tid = threadIdx.x;
    const int lane = tid & 31, wave = tid >> 5;
    const int wm = (wave & 3) * 32;
    const int wn = (wave >> 2) * 64;
    const int rowBase = blockIdx.y * 128;
    const int colBase = blockIdx.x * 128;

    v8f c[2][4];
    #pragma unroll
    for (int i = 0; i < 2; ++i)
        #pragma unroll
        for (int j = 0; j < 4; ++j)
            c[i][j] = (v8f){0.f, 0.f, 0.f, 0.f, 0.f, 0.f, 0.f, 0.f};

    mma_wave_tile(xb, wg1t, DIN, rowBase + wm, colBase + wn, lane, c);

    const int hq = (lane >> 4) << 3;  // C layout: M = p + 8*(lane>=16)
    #pragma unroll
    for (int i = 0; i < 2; ++i) {
        #pragma unroll
        for (int p = 0; p < 8; ++p) {
            const int n = rowBase + wm + i * 16 + hq + p;
            #pragma unroll
            for (int j = 0; j < 4; ++j) {
                const int o = colBase + wn + j * 16 + (lane & 15);
                float val = c[i][j][p] + bg1[o];
                val = val > 0.f ? val : 0.f;
                h[(size_t)n * HGATE + o] = (bf16_t)val;
            }
        }
    }
}

// ---------------------------------------------------------------------------
// Kernel 3: per-token gate logits, top-2 + softmax, and rank-16 projections
//   t_all[n][k*16+r] = gate_k * S[e_k][r] * dot(x[n,:], V[e_k][r,:])
// One wave32 per token, 4 tokens per 128-thread block.
// ---------------------------------------------------------------------------
__global__ __launch_bounds__(128) void gate_topk_t(
        const float* __restrict__ x, const bf16_t* __restrict__ h,
        const float* __restrict__ Wg2, const float* __restrict__ bg2,
        const float* __restrict__ S_all, const float* __restrict__ V_all,
        float* __restrict__ t_all, int* __restrict__ top_idx) {
    const int lane = threadIdx.x & 31;
    const int n = blockIdx.x * 4 + (threadIdx.x >> 5);

    // --- logits: lane = (expert e = lane&7, chunk = lane>>3 of 64) ---
    const int e = lane & 7;
    const int ch = lane >> 3;
    const bf16_t* hrow = h + (size_t)n * HGATE;
    float part = 0.f;
    for (int i = ch * 64; i < ch * 64 + 64; ++i)
        part += (float)hrow[i] * Wg2[i * NE + e];
    part += __shfl_down(part, 16, 32);
    part += __shfl_down(part, 8, 32);   // lanes 0..7 hold full sums

    // --- top-2 + softmax, replicated on all lanes via broadcasts ---
    float l0 = -1e30f, l1 = -1e30f;
    int e0 = 0, e1 = 0;
    #pragma unroll
    for (int q = 0; q < NE; ++q) {
        const float lv = __shfl(part, q, 32) + bg2[q];
        if (lv > l0)      { l1 = l0; e1 = e0; l0 = lv; e0 = q; }
        else if (lv > l1) { l1 = lv; e1 = q; }
    }
    float g1 = __expf(l1 - l0);
    const float inv = 1.f / (1.f + g1);
    const float g0 = inv;
    g1 *= inv;

    // --- rank-16 dots: lane -> (k = lane>>4, r = lane&15) ---
    const int k = lane >> 4;
    const int r = lane & 15;
    const int ee = k ? e1 : e0;
    const float gg = k ? g1 : g0;
    const v4f* xr = (const v4f*)(x + (size_t)n * DIN);
    const v4f* vr = (const v4f*)(V_all + ((size_t)ee * RK + r) * DIN);
    float acc = 0.f;
    for (int q = 0; q < DIN / 4; ++q) {
        const v4f a = xr[q], b = vr[q];
        acc += a.x * b.x + a.y * b.y + a.z * b.z + a.w * b.w;
    }
    t_all[n * 32 + lane] = acc * S_all[ee * RK + r] * gg;
    if (lane == 0) { top_idx[2 * n] = e0; top_idx[2 * n + 1] = e1; }
}

// ---------------------------------------------------------------------------
// Kernel 4: out = x @ Wm^T + bias + expert epilogue (fused).
// Block tile 128x128, 8 waves, wave tile 32x64 (2x4 WMMA).
// Dynamic LDS (~81 KB): t rows (16 KB) + U[8][128][16] (64 KB) + idx (1 KB).
// ---------------------------------------------------------------------------
#define MAIN_SMEM_FLOATS (128 * 32 + NE * 128 * RK)          // sT + sU
#define MAIN_SMEM_BYTES  (MAIN_SMEM_FLOATS * 4 + 256 * 4)    // + sIdx

__global__ __launch_bounds__(256) void moe_main_gemm(
        const bf16_t* __restrict__ xb, const bf16_t* __restrict__ wmb,
        const float* __restrict__ bias, const float* __restrict__ U_all,
        const float* __restrict__ t_all, const int* __restrict__ top_idx,
        float* __restrict__ out) {
    extern __shared__ char smem[];
    float* sT   = (float*)smem;                 // [128][32]
    float* sU   = sT + 128 * 32;                // [8][128][16]
    int*   sIdx = (int*)(sU + NE * 128 * RK);   // [128][2]

    const int tid = threadIdx.x;
    const int lane = tid & 31, wave = tid >> 5;
    const int wm = (wave & 3) * 32;
    const int wn = (wave >> 2) * 64;
    const int rowBase = blockIdx.y * 128;
    const int colBase = blockIdx.x * 128;

    v8f c[2][4];
    #pragma unroll
    for (int i = 0; i < 2; ++i)
        #pragma unroll
        for (int j = 0; j < 4; ++j)
            c[i][j] = (v8f){0.f, 0.f, 0.f, 0.f, 0.f, 0.f, 0.f, 0.f};

    mma_wave_tile(xb, wmb, DIN, rowBase + wm, colBase + wn, lane, c);

    // --- stage epilogue operands into LDS ---
    {
        const v4f* tsrc = (const v4f*)(t_all + (size_t)rowBase * 32);
        v4f* tdst = (v4f*)sT;
        #pragma unroll
        for (int q = 0; q < 4; ++q)                 // 128 rows x 32 floats
            tdst[tid * 4 + q] = tsrc[tid * 4 + q];

        sIdx[tid] = top_idx[rowBase * 2 + tid];     // 128 rows x 2 ints

        v4f* udst = (v4f*)sU;
        #pragma unroll
        for (int q = 0; q < 16; ++q) {              // 8 experts x 128 cols x 16 r
            const int flat = tid * 64 + q * 4;
            const int ue = flat >> 11;              // 128*16 floats per expert
            const int rem = flat & 2047;
            const int ol = rem >> 4;
            const int rr = rem & 15;                // always multiple of 4
            udst[flat >> 2] = *(const v4f*)(U_all +
                ((size_t)ue * DOUT + colBase + ol) * RK + rr);
        }
    }
    __syncthreads();

    // --- fused epilogue: C + bias + sum_{k,r} t * U ---
    const int hq = (lane >> 4) << 3;
    #pragma unroll
    for (int i = 0; i < 2; ++i) {
        #pragma unroll
        for (int p = 0; p < 8; ++p) {
            const int m = wm + i * 16 + hq + p;     // block-local row
            const int ne0 = sIdx[2 * m];
            const int ne1 = sIdx[2 * m + 1];
            const v4f* tv = (const v4f*)(sT + m * 32);
            const size_t n = (size_t)(rowBase + m);
            #pragma unroll
            for (int j = 0; j < 4; ++j) {
                const int ol = wn + j * 16 + (lane & 15);   // block-local col
                const int o = colBase + ol;
                float val = c[i][j][p] + bias[o];
                const v4f* u0 = (const v4f*)(sU + (ne0 * 128 + ol) * RK);
                const v4f* u1 = (const v4f*)(sU + (ne1 * 128 + ol) * RK);
                #pragma unroll
                for (int q = 0; q < 4; ++q) {
                    const v4f t0 = tv[q],     a0 = u0[q];
                    const v4f t1 = tv[4 + q], a1 = u1[q];
                    val += t0.x * a0.x + t0.y * a0.y + t0.z * a0.z + t0.w * a0.w;
                    val += t1.x * a1.x + t1.y * a1.y + t1.z * a1.z + t1.w * a1.w;
                }
                out[n * DOUT + o] = val;
            }
        }
    }
}

// ---------------------------------------------------------------------------
// Host launcher. Workspace layout (all offsets 256B-aligned, ~13.1 MB total):
//   [0)          x_bf   : 4096*1024 bf16 = 8 MB
//   [8388608)    wm_bf  : 1024*1024 bf16 = 2 MB
//   [10485760)   wg1t   :  256*1024 bf16 = 512 KB  (transposed Wg1)
//   [11010048)   h_bf   : 4096*256  bf16 = 2 MB
//   [13107200)   t_all  : 4096*32   f32  = 512 KB
//   [13631488)   topIdx : 4096*2    i32  = 32 KB
// ---------------------------------------------------------------------------
extern "C" void kernel_launch(void* const* d_in, const int* in_sizes, int n_in,
                              void* d_out, int out_size, void* d_ws, size_t ws_size,
                              hipStream_t stream) {
    const float* x     = (const float*)d_in[0];
    const float* wmain = (const float*)d_in[1];
    const float* bias  = (const float*)d_in[2];
    const float* U_all = (const float*)d_in[3];
    const float* S_all = (const float*)d_in[4];
    const float* V_all = (const float*)d_in[5];
    const float* Wg1   = (const float*)d_in[6];
    const float* bg1   = (const float*)d_in[7];
    const float* Wg2   = (const float*)d_in[8];
    const float* bg2   = (const float*)d_in[9];
    float* out = (float*)d_out;

    char* ws = (char*)d_ws;
    bf16_t* xb    = (bf16_t*)(ws);
    bf16_t* wmb   = (bf16_t*)(ws + 8388608);
    bf16_t* wg1t  = (bf16_t*)(ws + 10485760);
    bf16_t* hbf   = (bf16_t*)(ws + 11010048);
    float*  t_all = (float*)(ws + 13107200);
    int*    tidx  = (int*)(ws + 13631488);

    convert_kernel<<<dim3(1024), dim3(256), 0, stream>>>(x, wmain, Wg1, xb, wmb, wg1t);

    gate_hidden_gemm<<<dim3(HGATE / 128, NTOK / 128), dim3(256), 0, stream>>>(
        xb, wg1t, bg1, hbf);

    gate_topk_t<<<dim3(NTOK / 4), dim3(128), 0, stream>>>(
        x, hbf, Wg2, bg2, S_all, V_all, t_all, tidx);

    moe_main_gemm<<<dim3(DOUT / 128, NTOK / 128), dim3(256), MAIN_SMEM_BYTES, stream>>>(
        xb, wmb, bias, U_all, t_all, tidx, out);
}
